// MambaBlock_17944373363200
// MI455X (gfx1250) — compile-verified, matrix operand order checked by reading, not run on hardware
//
#include <hip/hip_runtime.h>
#include <hip/hip_bf16.h>
#include <math.h>

typedef __attribute__((ext_vector_type(16))) _Float16 v16h;
typedef __attribute__((ext_vector_type(8)))  float    v8f;

#define MROWS 4096   // B * ENC_IN
#define SEQ   512
#define CIN   256
#define PRED  96

__device__ __forceinline__ float silu_f(float x)    { return x / (1.0f + __expf(-x)); }
__device__ __forceinline__ float softplus_f(float x){ return (x > 20.0f) ? x : log1pf(__expf(x)); }

// ---------------------------------------------------------------------------
// RevIN: per (batch, channel) mean / stdev over seq, write normalized rows
// transposed into XN (4096 x 512), save mean/stdev for final de-norm.
// Block = 256 threads = (8 seq-slices x 32 channels); grid = (16 batches, 8 cgroups)
// ---------------------------------------------------------------------------
__global__ void revin_kernel(const float* __restrict__ x,
                             const float* __restrict__ rw,
                             const float* __restrict__ rb,
                             float* __restrict__ xn,
                             float* __restrict__ meanBuf,
                             float* __restrict__ stdBuf)
{
    __shared__ float s_sum[256], s_ssq[256], s_mean[32], s_inv[32], s_sd[32];
    int b    = blockIdx.x;                  // 0..15
    int cg   = blockIdx.y;                  // 0..7
    int cidx = threadIdx.x & 31;            // channel within group (coalesced)
    int sidx = threadIdx.x >> 5;            // seq slice 0..7
    int c    = cg * 32 + cidx;
    const float* px = x + (size_t)b * SEQ * CIN + c;
    float s = 0.f, ss = 0.f;
    for (int t = sidx * 64; t < sidx * 64 + 64; ++t) {
        float v = px[(size_t)t * CIN];
        s += v; ss += v * v;
    }
    s_sum[threadIdx.x] = s; s_ssq[threadIdx.x] = ss;
    __syncthreads();
    if (sidx == 0) {
        float ts = 0.f, tq = 0.f;
        for (int j = 0; j < 8; ++j) { ts += s_sum[j * 32 + cidx]; tq += s_ssq[j * 32 + cidx]; }
        float mean = ts * (1.0f / SEQ);
        float var  = tq * (1.0f / SEQ) - mean * mean;
        float sd   = sqrtf(var);
        s_mean[cidx] = mean; s_inv[cidx] = 1.0f / sd; s_sd[cidx] = sd;
    }
    __syncthreads();
    float mean = s_mean[cidx], inv = s_inv[cidx];
    int row = b * CIN + c;
    if (sidx == 0) { meanBuf[row] = mean; stdBuf[row] = s_sd[cidx]; }
    float w = rw[c], bb = rb[c];
    float* pr = xn + (size_t)row * SEQ;
    for (int t = sidx * 64; t < sidx * 64 + 64; ++t)
        pr[t] = (px[(size_t)t * CIN] - mean) * inv * w + bb;
}

// ---------------------------------------------------------------------------
// WMMA GEMM:  C[M,N] = A[M,K] * W[N,K]^T (+bias[N]) (+C0[M,N])
// A, W are f32 in memory (16B-aligned rows), loaded as float4 and converted
// in-register to f16; f32 accumulate via v_wmma_f32_16x16x32_f16.
// M = 4096 (multiple of 32), K multiple of 32.
// GUARD=false: N multiple of 64 (pure load->cvt->wmma hot loop).
// GUARD=true : ragged N, branch-free mask on the B fragment + store guard.
// Block = 256 threads = 8 waves -> 32x64 output tile per block.
// ---------------------------------------------------------------------------
template <bool GUARD>
__global__ void wmma_gemm_nt(const float* __restrict__ A, int lda,
                             const float* __restrict__ W, int ldw,
                             const float* __restrict__ bias,
                             const float* __restrict__ C0, int ldc0,
                             float* __restrict__ C, int ldc,
                             int N, int K)
{
    const int lane = threadIdx.x & 31;
    const int wave = threadIdx.x >> 5;
    const int hl   = lane >> 4;     // K-phase: lanes<16 -> K{0-7,16-23}; >=16 -> K{8-15,24-31}
    const int r    = lane & 15;     // M index in A-frag / N index in B-frag & C/D
    const int tileM = blockIdx.y * 32 + (wave >> 2) * 16;
    const int tileN = blockIdx.x * 64 + (wave & 3) * 16;

    const int   rowW   = tileN + r;
    const bool  wvalid = GUARD ? (rowW < N) : true;
    const float wm     = wvalid ? 1.0f : 0.0f;   // branch-free N-guard for B frag

    const float4* pa4 = (const float4*)(A + (size_t)(tileM + r) * lda) + hl * 2;
    const float4* pw4 = (const float4*)(W + (size_t)(wvalid ? rowW : 0) * ldw) + hl * 2;

    v8f acc = {};
#pragma unroll 2
    for (int k0 = 0; k0 < K; k0 += 32, pa4 += 8, pw4 += 8) {
        float4 a0 = pa4[0], a1 = pa4[1], a2 = pa4[4], a3 = pa4[5];
        float4 w0 = pw4[0], w1 = pw4[1], w2 = pw4[4], w3 = pw4[5];
        if (GUARD) {
            w0.x *= wm; w0.y *= wm; w0.z *= wm; w0.w *= wm;
            w1.x *= wm; w1.y *= wm; w1.z *= wm; w1.w *= wm;
            w2.x *= wm; w2.y *= wm; w2.z *= wm; w2.w *= wm;
            w3.x *= wm; w3.y *= wm; w3.z *= wm; w3.w *= wm;
        }
        v16h a, bfr;
        a[0]  = (_Float16)a0.x; a[1]  = (_Float16)a0.y; a[2]  = (_Float16)a0.z; a[3]  = (_Float16)a0.w;
        a[4]  = (_Float16)a1.x; a[5]  = (_Float16)a1.y; a[6]  = (_Float16)a1.z; a[7]  = (_Float16)a1.w;
        a[8]  = (_Float16)a2.x; a[9]  = (_Float16)a2.y; a[10] = (_Float16)a2.z; a[11] = (_Float16)a2.w;
        a[12] = (_Float16)a3.x; a[13] = (_Float16)a3.y; a[14] = (_Float16)a3.z; a[15] = (_Float16)a3.w;
        bfr[0]  = (_Float16)w0.x; bfr[1]  = (_Float16)w0.y; bfr[2]  = (_Float16)w0.z; bfr[3]  = (_Float16)w0.w;
        bfr[4]  = (_Float16)w1.x; bfr[5]  = (_Float16)w1.y; bfr[6]  = (_Float16)w1.z; bfr[7]  = (_Float16)w1.w;
        bfr[8]  = (_Float16)w2.x; bfr[9]  = (_Float16)w2.y; bfr[10] = (_Float16)w2.z; bfr[11] = (_Float16)w2.w;
        bfr[12] = (_Float16)w3.x; bfr[13] = (_Float16)w3.y; bfr[14] = (_Float16)w3.z; bfr[15] = (_Float16)w3.w;
        acc = __builtin_amdgcn_wmma_f32_16x16x32_f16(false, a, false, bfr,
                                                     (short)0, acc, false, false);
    }

    const int col = tileN + r;
    if (!GUARD || col < N) {
        float bv = bias ? bias[col] : 0.0f;
#pragma unroll
        for (int i = 0; i < 8; ++i) {
            int row = tileM + hl * 8 + i;     // C/D layout: lane<16 -> M=i, lane>=16 -> M=8+i
            float v = acc[i] + bv;
            if (C0) v += C0[(size_t)row * ldc0 + col];
            C[(size_t)row * ldc + col] = v;
        }
    }
}

// ---------------------------------------------------------------------------
// Mamba (L=1) pieces: causal conv over length-1 seq == w[:,3]*x + b, then silu
// ---------------------------------------------------------------------------
__global__ void conv_silu_kernel(const float* __restrict__ xz,
                                 const float* __restrict__ convw,
                                 const float* __restrict__ convb,
                                 float* __restrict__ xh, int di)
{
    int idx = blockIdx.x * blockDim.x + threadIdx.x;  // row*di + d
    int row = idx / di, d = idx - row * di;
    float v = xz[(size_t)row * (2 * di) + d];
    xh[idx] = silu_f(convw[d * 4 + 3] * v + convb[d]);
}

// L=1 scan collapses: y = xh * (softplus(dt@dtw+b) * (B.C) + D) * silu(z)
__global__ void mamba_finish_kernel(const float* __restrict__ dbc, int lddbc, int dtr,
                                    const float* __restrict__ xh,
                                    const float* __restrict__ xz,   // z half
                                    const float* __restrict__ dtw,  // (di, dtr)
                                    const float* __restrict__ dtb,
                                    const float* __restrict__ Dp,
                                    float* __restrict__ y, int di)
{
    int idx = blockIdx.x * blockDim.x + threadIdx.x;
    int row = idx / di, d = idx - row * di;
    const float* pd = dbc + (size_t)row * lddbc;
    float acc = dtb[d];
    for (int r = 0; r < dtr; ++r) acc += pd[r] * dtw[d * dtr + r];
    float delta = softplus_f(acc);
    float bc = 0.f;
#pragma unroll
    for (int n = 0; n < 16; ++n) bc += pd[dtr + n] * pd[dtr + 16 + n];
    float xv = xh[idx];
    float z  = xz[(size_t)row * (2 * di) + di + d];
    y[idx] = xv * (delta * bc + Dp[d]) * silu_f(z);
}

// ---------------------------------------------------------------------------
// m4: real selective scan, d_model=1, d_inner=2, d_state=16, dt_rank=1.
// One wave per row; lane = d*16 + n holds state h[d][n] (2x16 == wave32).
// Epilogue fuses the x4+x3 add and writes into CAT columns [256,512).
// ---------------------------------------------------------------------------
__global__ void mamba_scan1_kernel(const float* __restrict__ U,     // (M, L)
                                   const float* __restrict__ addv,  // (M, L) = X3
                                   float* __restrict__ outp,        // CAT
                                   int L, int outStride, int outOfs,
                                   const float* __restrict__ inw,   // (4,1)
                                   const float* __restrict__ cw,    // (2,4)
                                   const float* __restrict__ cb,    // (2)
                                   const float* __restrict__ xpw,   // (33,2)
                                   const float* __restrict__ dtw,   // (2,1)
                                   const float* __restrict__ dtb,   // (2)
                                   const float* __restrict__ alog,  // (2,16)
                                   const float* __restrict__ Dp,    // (2)
                                   const float* __restrict__ ow)    // (1,2)
{
    int lane = threadIdx.x & 31;
    int wave = threadIdx.x >> 5;
    int row  = blockIdx.x * 8 + wave;
    int d = lane >> 4, n = lane & 15;

    float inw0 = inw[0], inw1 = inw[1], inw2 = inw[2], inw3 = inw[3];
    float cw00 = cw[0], cw01 = cw[1], cw02 = cw[2], cw03 = cw[3];
    float cw10 = cw[4], cw11 = cw[5], cw12 = cw[6], cw13 = cw[7];
    float cb0 = cb[0], cb1 = cb[1];
    float xpd0 = xpw[0], xpd1 = xpw[1];
    float xpB0 = xpw[(1 + n) * 2],  xpB1 = xpw[(1 + n) * 2 + 1];
    float xpC0 = xpw[(17 + n) * 2], xpC1 = xpw[(17 + n) * 2 + 1];
    float dtwd = dtw[d], dtbd = dtb[d];
    float An = -__expf(alog[lane]);          // A_log flat index == lane
    float Dd = Dp[d], ow0 = ow[0], ow1 = ow[1];

    float b00 = 0, b01 = 0, b02 = 0, b10 = 0, b11 = 0, b12 = 0;  // conv history
    float h = 0.f;
    const float* pu = U    + (size_t)row * L;
    const float* pa = addv + (size_t)row * L;
    float*       po = outp + (size_t)row * outStride + outOfs;

    for (int t = 0; t < L; ++t) {
        float u  = pu[t];
        float xr0 = inw0 * u, xr1 = inw1 * u;     // in_proj: xh-half
        float z0  = inw2 * u, z1  = inw3 * u;     // in_proj: z-half
        float c0 = cw00 * b00 + cw01 * b01 + cw02 * b02 + cw03 * xr0 + cb0;
        float c1 = cw10 * b10 + cw11 * b11 + cw12 * b12 + cw13 * xr1 + cb1;
        b00 = b01; b01 = b02; b02 = xr0;
        b10 = b11; b11 = b12; b12 = xr1;
        float xh0 = silu_f(c0), xh1 = silu_f(c1);
        float dts = xpd0 * xh0 + xpd1 * xh1;
        float Bn  = xpB0 * xh0 + xpB1 * xh1;
        float Cn  = xpC0 * xh0 + xpC1 * xh1;
        float xhd = d ? xh1 : xh0;
        float zd  = d ? z1 : z0;
        float delta = softplus_f(dts * dtwd + dtbd);
        float dA = __expf(delta * An);
        h = dA * h + delta * Bn * xhd;
        float p = h * Cn;
        p += __shfl_xor(p, 1, 32);
        p += __shfl_xor(p, 2, 32);
        p += __shfl_xor(p, 4, 32);
        p += __shfl_xor(p, 8, 32);                 // sum over n within half-wave
        float yd = (p + Dd * xhd) * silu_f(zd);
        float yo = __shfl_xor(yd, 16, 32);         // other d's y
        float out = d ? (ow0 * yo + ow1 * yd) : (ow0 * yd + ow1 * yo);
        if (lane == 0) po[t] = out + pa[t];
    }
}

__global__ void h3_kernel(const float* __restrict__ h2, const float* __restrict__ x2,
                          float* __restrict__ h3, int nelem)
{
    int i = blockIdx.x * blockDim.x + threadIdx.x;
    if (i < nelem) h3[i] = 2.0f * h2[i] + x2[i];   // h + x_res2 + x2 with h==x_res2
}

__global__ void denorm_kernel(const float* __restrict__ outm,  // (M, 96)
                              const float* __restrict__ rw,
                              const float* __restrict__ rb,
                              const float* __restrict__ meanBuf,
                              const float* __restrict__ stdBuf,
                              float* __restrict__ dout)          // (16, 96, 256)
{
    int b = blockIdx.x / PRED;
    int t = blockIdx.x - b * PRED;
    int c = threadIdx.x;
    int row = b * CIN + c;
    float v = outm[(size_t)row * PRED + t];
    v = (v - rb[c]) / (rw[c] + 1e-10f) * stdBuf[row] + meanBuf[row];
    dout[((size_t)b * PRED + t) * CIN + c] = v;
}

// ---------------------------------------------------------------------------
extern "C" void kernel_launch(void* const* d_in, const int* in_sizes, int n_in,
                              void* d_out, int out_size, void* d_ws, size_t ws_size,
                              hipStream_t stream)
{
    (void)in_sizes; (void)n_in; (void)out_size; (void)ws_size;
    const float* x   = (const float*)d_in[0];
    const float* rw  = (const float*)d_in[1];
    const float* rb  = (const float*)d_in[2];
    const float* l1w = (const float*)d_in[3];
    const float* l1b = (const float*)d_in[4];
    const float* l2w = (const float*)d_in[5];
    const float* l2b = (const float*)d_in[6];
    const float* l3w = (const float*)d_in[7];
    const float* l3b = (const float*)d_in[8];
    const float* l4w = (const float*)d_in[9];
    const float* l4b = (const float*)d_in[10];

    struct MP { const float *inp,*cw,*cb,*xp,*dtw,*dtb,*alog,*D,*ow; };
    auto getmp = [&](int b) {
        MP m;
        m.inp  = (const float*)d_in[b + 0]; m.cw  = (const float*)d_in[b + 1];
        m.cb   = (const float*)d_in[b + 2]; m.xp  = (const float*)d_in[b + 3];
        m.dtw  = (const float*)d_in[b + 4]; m.dtb = (const float*)d_in[b + 5];
        m.alog = (const float*)d_in[b + 6]; m.D   = (const float*)d_in[b + 7];
        m.ow   = (const float*)d_in[b + 8];
        return m;
    };
    // dict order: 0..10 top-level, m1 @11 (dead code -> skipped), m2 @20, m3 @29, m4 @38
    MP m2p = getmp(20), m3p = getmp(29), m4p = getmp(38);

    // ---- workspace layout (lifetime-based reuse, ~45 MB) ----
    char* ws = (char*)d_ws;
    const size_t MB = 1024 * 1024;
    float* meanB = (float*)(ws + 0);
    float* stdB  = (float*)(ws + 16384);
    float* XN    = (float*)(ws + 32768);            // 8MB; dead after lin1
    float* Y3    = XN;                              // reuse (written after XN dead)
    float* X2    = XN;                              // reuse (written after Y3 dead)
    float* H3    = (float*)(ws + 32768 + 2 * MB);   // reuse tail of XN region
    float* H1    = (float*)(ws + 32768 + 8 * MB);   // 4MB, long-lived (x_res1)
    float* XZ3   = (float*)(ws + 32768 + 12 * MB);  // 16MB
    float* XZ2   = XZ3;                             // reuse front 8MB
    float* CAT   = (float*)(ws + 32768 + 20 * MB);  // 8MB (XZ3 back half, XZ3 dead)
    float* XH3   = (float*)(ws + 32768 + 28 * MB);  // 8MB
    float* XH2   = XH3;                             // reuse front 4MB
    float* Y2    = (float*)(ws + 32768 + 32 * MB);  // 4MB (XH3 back half)
    float* DBC   = (float*)(ws + 32768 + 36 * MB);  // 1MB (DBC3 then DBC2)
    float* X3    = (float*)(ws + 32768 + 37 * MB);  // 4MB
    float* H2    = (float*)(ws + 32768 + 41 * MB);  // 2MB (x_res2)
    float* OUTB  = (float*)(ws + 32768 + 43 * MB);  // 1.5MB

    auto gemm = [&](const float* A, int lda, const float* W, int ldw,
                    const float* bias, const float* C0, int ldc0,
                    float* C, int ldc, int N, int K) {
        dim3 g((N + 63) / 64, MROWS / 32);
        if ((N & 63) == 0)
            wmma_gemm_nt<false><<<g, 256, 0, stream>>>(A, lda, W, ldw, bias, C0, ldc0, C, ldc, N, K);
        else
            wmma_gemm_nt<true><<<g, 256, 0, stream>>>(A, lda, W, ldw, bias, C0, ldc0, C, ldc, N, K);
    };

    // 1) RevIN
    revin_kernel<<<dim3(16, 8), 256, 0, stream>>>(x, rw, rb, XN, meanB, stdB);
    // 2) lin1: H1 = XN @ l1w^T + b        (4096x256, K=512)
    gemm(XN, SEQ, l1w, SEQ, l1b, nullptr, 0, H1, 256, 256, SEQ);
    // 3-7) m3 (d_model=256, L=1)
    gemm(H1, 256, m3p.inp, 256, nullptr, nullptr, 0, XZ3, 1024, 1024, 256);
    conv_silu_kernel<<<(MROWS * 512) / 256, 256, 0, stream>>>(XZ3, m3p.cw, m3p.cb, XH3, 512);
    gemm(XH3, 512, m3p.xp, 512, nullptr, nullptr, 0, DBC, 48, 48, 512);
    mamba_finish_kernel<<<(MROWS * 512) / 256, 256, 0, stream>>>(DBC, 48, 16, XH3, XZ3,
                                                                 m3p.dtw, m3p.dtb, m3p.D, Y3, 512);
    gemm(Y3, 512, m3p.ow, 512, nullptr, nullptr, 0, X3, 256, 256, 512);
    // 8) m4 scan over 256 features; fuses (+X3) and writes CAT[:,256:512)
    mamba_scan1_kernel<<<MROWS / 8, 256, 0, stream>>>(H1, X3, CAT, 256, 512, 256,
        m4p.inp, m4p.cw, m4p.cb, m4p.xp, m4p.dtw, m4p.dtb, m4p.alog, m4p.D, m4p.ow);
    // 9) lin2: H2 = H1 @ l2w^T + b        (4096x128, K=256)
    gemm(H1, 256, l2w, 256, l2b, nullptr, 0, H2, 128, 128, 256);
    // 10-14) m2 (d_model=128, L=1)
    gemm(H2, 128, m2p.inp, 128, nullptr, nullptr, 0, XZ2, 512, 512, 128);
    conv_silu_kernel<<<(MROWS * 256) / 256, 256, 0, stream>>>(XZ2, m2p.cw, m2p.cb, XH2, 256);
    gemm(XH2, 256, m2p.xp, 256, nullptr, nullptr, 0, DBC, 40, 40, 256);
    mamba_finish_kernel<<<(MROWS * 256) / 256, 256, 0, stream>>>(DBC, 40, 8, XH2, XZ2,
                                                                 m2p.dtw, m2p.dtb, m2p.D, Y2, 256);
    gemm(Y2, 256, m2p.ow, 256, nullptr, nullptr, 0, X2, 128, 128, 256);
    // 15) H3 = 2*H2 + X2
    h3_kernel<<<(MROWS * 128) / 256, 256, 0, stream>>>(H2, X2, H3, MROWS * 128);
    // 16) lin3 + x_res1, written into CAT[:,0:256)
    gemm(H3, 128, l3w, 128, l3b, H1, 256, CAT, 512, 256, 128);
    // 17) lin4 on concat (4096x512 -> 4096x96)
    gemm(CAT, 512, l4w, 512, l4b, nullptr, 0, OUTB, 96, 96, 512);
    // 18) de-norm + transpose to (16, 96, 256)
    denorm_kernel<<<16 * PRED, 256, 0, stream>>>(OUTB, rw, rb, meanB, stdB, (float*)d_out);
}
